// MultiHeadSelfAttention_23476291240173
// MI455X (gfx1250) — matrix operand order repaired and checked by reading, not compiled, not run on hardware
//
#include <hip/hip_runtime.h>

#define DEV __device__ __forceinline__

typedef __attribute__((ext_vector_type(16))) __bf16       v16bf;
typedef __attribute__((ext_vector_type(8)))  float        v8f;
typedef __attribute__((ext_vector_type(4)))  unsigned int u32x4;
typedef int v4i __attribute__((vector_size(16)));   // matches builtin param type

union FragU { v16bf v; unsigned short u[16]; u32x4 q[2]; };

#define WMMA_BF16(a, b, c) \
  __builtin_amdgcn_wmma_f32_16x16x32_bf16(false, (a), false, (b), (short)0, (c), false, false)

#ifndef __has_builtin
#define __has_builtin(x) 0
#endif
#if __has_builtin(__builtin_amdgcn_global_load_async_to_lds_b128) && \
    __has_builtin(__builtin_amdgcn_s_wait_asynccnt)
#define HAVE_ASYNC_LDS 1
#else
#define HAVE_ASYNC_LDS 0
#endif

#define AS1 __attribute__((address_space(1)))
#define AS3 __attribute__((address_space(3)))

#if HAVE_ASYNC_LDS
// CDNA5 async global->LDS copy (no VGPR round trip, tracked by ASYNCcnt)
DEV void cp_async_b128(void* lds, const void* g) {
  __builtin_amdgcn_global_load_async_to_lds_b128((AS1 v4i*)g, (AS3 v4i*)lds, 0, 0);
}
DEV void cp_async_wait() { __builtin_amdgcn_s_wait_asynccnt(0); }
#else
DEV void cp_async_b128(void* lds, const void* g) { *(u32x4*)lds = *(const u32x4*)g; }
DEV void cp_async_wait() {}
#endif

DEV unsigned short f2bf(float f) {  // round-to-nearest-even f32 -> bf16
  unsigned int u = __float_as_uint(f);
  return (unsigned short)((u + 0x7FFFu + ((u >> 16) & 1u)) >> 16);
}

// A-fragment (16x32 bf16): lane l holds row m=l%16;
// elems 0..15 = K[kOff + (l/16)*16 ..] (contiguous)
DEV v16bf frag_A_lds(const unsigned short* base, int stride, int kOff) {
  int l = threadIdx.x & 31;
  const unsigned short* row = base + (l & 15) * stride + kOff + (l >> 4) * 16;
  FragU r;
  r.q[0] = *(const u32x4*)(row);
  r.q[1] = *(const u32x4*)(row + 8);
  return r.v;
}

// B-fragment (32x16 bf16) from an LDS tile stored TRANSPOSED as [n][k]:
// lane l holds col n=l%16, K = (l/16)*16 .. +15 (contiguous).
DEV v16bf frag_B_lds(const unsigned short* base, int stride, int nOff, int kOff) {
  int l = threadIdx.x & 31;
  const unsigned short* row = base + (nOff + (l & 15)) * stride + kOff + (l >> 4) * 16;
  FragU r;
  r.q[0] = ((const u32x4*)row)[0];
  r.q[1] = ((const u32x4*)row)[1];
  return r.v;
}

// ---------------------------------------------------------------- converts
__global__ void cvt_bf16(const float* __restrict__ src, unsigned short* __restrict__ dst, int n) {
  int i = blockIdx.x * blockDim.x + threadIdx.x;
  if (i < n) dst[i] = f2bf(src[i]);
}

// W [768 k][2304 c] f32 -> wbT [2304 c][768 k] bf16, 32x32 LDS tile transpose
__global__ __launch_bounds__(256) void cvt_w_transpose(const float* __restrict__ W,
                                                       unsigned short* __restrict__ wbT) {
  __shared__ unsigned short tile[32][33];
  int kt = blockIdx.x * 32;                       // 768/32 = 24
  int ct = blockIdx.y * 32;                       // 2304/32 = 72
  int tx = threadIdx.x & 31, ty = threadIdx.x >> 5;
#pragma unroll
  for (int r = 0; r < 4; ++r)
    tile[ty + r * 8][tx] = f2bf(W[(size_t)(kt + ty + r * 8) * 2304 + ct + tx]);
  __syncthreads();
#pragma unroll
  for (int r = 0; r < 4; ++r)
    wbT[(size_t)(ct + ty + r * 8) * 768 + kt + tx] = tile[tx][ty + r * 8];
}

// ---------------------------------------------------------------- QKV GEMM
// [8192 x 2304] = xb[8192 x 768] * wbT^T + bias; scattered into per-head
// Qb/Kb [b,h,tok,hd] (Q pre-scaled by 1/8) and V TRANSPOSED Vt [b,h,hd,tok].
// Double-buffered async LDS staging; no unroll (single body -> accumulators
// coalesce into pinned WMMA tuples, no PHI copies / hazard nops).
__global__ __launch_bounds__(256, 1) void qkv_gemm(
    const unsigned short* __restrict__ xb, const unsigned short* __restrict__ wbT,
    const float* __restrict__ bias,
    unsigned short* __restrict__ Qb, unsigned short* __restrict__ Kb,
    unsigned short* __restrict__ Vt_g) {
  __shared__ __align__(16) unsigned short At[2][64][40];    // [m][k]
  __shared__ __align__(16) unsigned short Bt[2][256][40];   // [n][k]

  const int t = threadIdx.x, l = t & 31, w = t >> 5;
  const int m0 = blockIdx.x * 64;
  const int n0 = blockIdx.y * 256;
  const int m_sub = (w & 1) * 32;
  const int n_sub = (w >> 1) * 64;

  const int arow = t >> 2, acol = (t & 3) * 8;
  const unsigned short* ag = xb + (size_t)(m0 + arow) * 768 + acol;
  const unsigned short* bg = wbT + (size_t)(n0 + t) * 768;

  v8f c00 = {}, c01 = {}, c02 = {}, c03 = {};
  v8f c10 = {}, c11 = {}, c12 = {}, c13 = {};

  // prologue: issue tile 0 into buffer 0
  cp_async_b128(&At[0][arow][acol], ag);
  cp_async_b128(&Bt[0][t][0],  bg);
  cp_async_b128(&Bt[0][t][8],  bg + 8);
  cp_async_b128(&Bt[0][t][16], bg + 16);
  cp_async_b128(&Bt[0][t][24], bg + 24);

#pragma clang loop unroll(disable)
  for (int it = 0; it < 24; ++it) {
    const int buf = it & 1;
    cp_async_wait();       // my tile's loads have landed in LDS
    __syncthreads();       // everyone's tile ready; prev buffer free
    {  // branch-free prefetch: last iteration re-fetches tile 23 (L2 hit, unread)
      const int k0 = (it + 1 < 24 ? it + 1 : 23) * 32;
      cp_async_b128(&At[buf ^ 1][arow][acol], ag + k0);
      cp_async_b128(&Bt[buf ^ 1][t][0],  bg + k0);
      cp_async_b128(&Bt[buf ^ 1][t][8],  bg + k0 + 8);
      cp_async_b128(&Bt[buf ^ 1][t][16], bg + k0 + 16);
      cp_async_b128(&Bt[buf ^ 1][t][24], bg + k0 + 24);
    }
    // materialize all fragments, then the 8 WMMAs (short, simple live ranges)
    v16bf a0 = frag_A_lds(&At[buf][m_sub][0], 40, 0);
    v16bf a1 = frag_A_lds(&At[buf][m_sub + 16][0], 40, 0);
    v16bf b0 = frag_B_lds(&Bt[buf][0][0], 40, n_sub, 0);
    v16bf b1 = frag_B_lds(&Bt[buf][0][0], 40, n_sub + 16, 0);
    v16bf b2 = frag_B_lds(&Bt[buf][0][0], 40, n_sub + 32, 0);
    v16bf b3 = frag_B_lds(&Bt[buf][0][0], 40, n_sub + 48, 0);
    c00 = WMMA_BF16(a0, b0, c00);
    c10 = WMMA_BF16(a1, b0, c10);
    c01 = WMMA_BF16(a0, b1, c01);
    c11 = WMMA_BF16(a1, b1, c11);
    c02 = WMMA_BF16(a0, b2, c02);
    c12 = WMMA_BF16(a1, b2, c12);
    c03 = WMMA_BF16(a0, b3, c03);
    c13 = WMMA_BF16(a1, b3, c13);
  }

  const v8f* cc[2][4] = { { &c00, &c01, &c02, &c03 }, { &c10, &c11, &c12, &c13 } };
#pragma unroll
  for (int fm = 0; fm < 2; ++fm)
#pragma unroll
    for (int fn = 0; fn < 4; ++fn) {
      const v8f& cf = *cc[fm][fn];
      int rowBase = m0 + m_sub + fm * 16;
      int colBase = n0 + n_sub + fn * 16;
      int col = colBase + (l & 15);
      float bv = bias[col];
      int which = col / 768;               // 0=Q 1=K 2=V (uniform per frag)
      int rem = col - which * 768;
      int h = rem >> 6, hd = rem & 63;
      if (which == 2) {
        // V transposed: lane's 8 C rows = 8 consecutive toks -> one b128 store
        int row0 = rowBase + 8 * (l >> 4);
        int b = row0 >> 10, tok0 = row0 & 1023;
        FragU pk;
#pragma unroll
        for (int i = 0; i < 8; ++i) pk.u[i] = f2bf(cf[i] + bv);
        *(u32x4*)(Vt_g + ((size_t)(b * 12 + h) * 64 + hd) * 1024 + tok0) = pk.q[0];
      } else {
        unsigned short* dst = (which == 0) ? Qb : Kb;
        float sc = (which == 0) ? 0.125f : 1.0f;   // fold softmax scale into Q
#pragma unroll
        for (int i = 0; i < 8; ++i) {
          int row = rowBase + i + 8 * (l >> 4);
          int b = row >> 10, tok = row & 1023;
          dst[((size_t)(b * 12 + h) * 1024 + tok) * 64 + hd] =
              f2bf((cf[i] + bv) * sc);
        }
      }
    }
}

// ---------------------------------------------------------------- attention
// Transposed flash attention: S^T = K*Q^T, O^T = V^T*P^T.
// Block = (b,h) x 64 queries; 4 waves, each wave owns 16 queries (one/lane-col).
// Double-buffered async K/V staging, one barrier per key tile, no unroll.
__global__ __launch_bounds__(128, 1) void mha_attn(
    const unsigned short* __restrict__ Qb, const unsigned short* __restrict__ Kb,
    const unsigned short* __restrict__ Vt_g, float* __restrict__ out) {
  __shared__ __align__(16) unsigned short Kt[2][32][72];   // [key][hd]
  __shared__ __align__(16) unsigned short Vt[2][64][40];   // [hd][key]

  const int t = threadIdx.x, l = t & 31, w = t >> 5, half = l >> 4;
  const int q0 = blockIdx.x * 64;
  const int bh = blockIdx.y;                               // b*12 + h
  const size_t base = (size_t)bh * 1024 * 64;

  // staging addresses (per thread, fixed across tiles)
  const int skey = t >> 2, soff = (t & 3) * 16;
  const unsigned short* kg = Kb + base + (size_t)skey * 64 + soff;
  const int shd = t >> 1, skoff = (t & 1) * 16;
  const unsigned short* vg = Vt_g + base + (size_t)shd * 1024 + skoff;

  // Q^T as B-fragments: lane holds query q=l%16; 16 contiguous hd per half
  FragU qb0, qb1;
  {
    const unsigned short* qp =
        Qb + base + (size_t)(q0 + w * 16 + (l & 15)) * 64 + half * 16;
    qb0.q[0] = ((const u32x4*)qp)[0];
    qb0.q[1] = ((const u32x4*)qp)[1];
    qb1.q[0] = ((const u32x4*)(qp + 32))[0];
    qb1.q[1] = ((const u32x4*)(qp + 32))[1];
  }

  float rmax = -3.0e38f, rsum = 0.f;                       // one query per lane
  v8f o0 = {}, o1 = {}, o2 = {}, o3 = {};

  // prologue: issue key tile 0 into buffer 0
  cp_async_b128(&Kt[0][skey][soff],     kg);
  cp_async_b128(&Kt[0][skey][soff + 8], kg + 8);
  cp_async_b128(&Vt[0][shd][skoff],     vg);
  cp_async_b128(&Vt[0][shd][skoff + 8], vg + 8);

#pragma clang loop unroll(disable)
  for (int it = 0; it < 32; ++it) {
    const int buf = it & 1;
    cp_async_wait();
    __syncthreads();
    {  // branch-free prefetch: last iteration re-fetches tile 31 (L2 hit, unread)
      const int key0 = (it + 1 < 32 ? it + 1 : 31) * 32;
      cp_async_b128(&Kt[buf ^ 1][skey][soff],     kg + (size_t)key0 * 64);
      cp_async_b128(&Kt[buf ^ 1][skey][soff + 8], kg + (size_t)key0 * 64 + 8);
      cp_async_b128(&Vt[buf ^ 1][shd][skoff],     vg + key0);
      cp_async_b128(&Vt[buf ^ 1][shd][skoff + 8], vg + key0 + 8);
    }

    // S^T = K * Q^T : s0 keys = i+8*half, s1 keys = 16+i+8*half, query = l%16
    v8f s0 = {}, s1 = {};
    s0 = WMMA_BF16(frag_A_lds(&Kt[buf][0][0], 72, 0),  qb0.v, s0);
    s0 = WMMA_BF16(frag_A_lds(&Kt[buf][0][0], 72, 32), qb1.v, s0);
    s1 = WMMA_BF16(frag_A_lds(&Kt[buf][16][0], 72, 0),  qb0.v, s1);
    s1 = WMMA_BF16(frag_A_lds(&Kt[buf][16][0], 72, 32), qb1.v, s1);

    // Online softmax: in-lane reduction + ONE cross-half shuffle
    float lm = fmaxf(s0[0], s1[0]);
#pragma unroll
    for (int i = 1; i < 8; ++i) lm = fmaxf(lm, fmaxf(s0[i], s1[i]));
    lm = fmaxf(lm, __shfl_xor(lm, 16, 32));
    float nm = fmaxf(rmax, lm);
    float sc = __expf(rmax - nm);
    rmax = nm;
    rsum *= sc;
#pragma unroll
    for (int i = 0; i < 8; ++i) { o0[i] *= sc; o1[i] *= sc; o2[i] *= sc; o3[i] *= sc; }

    FragU p0, p1;
    float ls = 0.f;
#pragma unroll
    for (int i = 0; i < 8; ++i) {
      float e0 = __expf(s0[i] - nm);
      float e1 = __expf(s1[i] - nm);
      ls += e0 + e1;
      p0.u[i] = f2bf(e0);
      p1.u[i] = f2bf(e1);
    }
    ls += __shfl_xor(ls, 16, 32);
    rsum += ls;

    // P^T B-fragment [32k x 16q] via one cross-half packed exchange:
    // half0 needs keys 0..15  = [own p0 | partner p0]
    // half1 needs keys 16..31 = [partner p1 | own p1]
    u32x4 sel = half ? p0.q[0] : p1.q[0];   // send what the partner needs
    u32x4 Y;
    Y.x = (unsigned int)__shfl_xor((int)sel.x, 16, 32);
    Y.y = (unsigned int)__shfl_xor((int)sel.y, 16, 32);
    Y.z = (unsigned int)__shfl_xor((int)sel.z, 16, 32);
    Y.w = (unsigned int)__shfl_xor((int)sel.w, 16, 32);
    FragU pb;
    pb.q[0] = half ? Y : p0.q[0];
    pb.q[1] = half ? p1.q[0] : Y;

    // O^T += V^T * P^T  (V^T LDS tiles are ready-made A-fragments)
    o0 = WMMA_BF16(frag_A_lds(&Vt[buf][0][0],  40, 0), pb.v, o0);
    o1 = WMMA_BF16(frag_A_lds(&Vt[buf][16][0], 40, 0), pb.v, o1);
    o2 = WMMA_BF16(frag_A_lds(&Vt[buf][32][0], 40, 0), pb.v, o2);
    o3 = WMMA_BF16(frag_A_lds(&Vt[buf][48][0], 40, 0), pb.v, o3);
  }

  // Normalize + store: per lane 8 contiguous hd floats per chunk
  int b = bh / 12, h = bh - b * 12;
  int q = q0 + w * 16 + (l & 15);
  float inv = 1.0f / rsum;
  float* op = out + ((size_t)b * 1024 + q) * 768 + h * 64 + 8 * half;
#pragma unroll
  for (int i = 0; i < 8; ++i) {
    op[i]      = o0[i] * inv;   // hd = i + 8*half + 16*chunk
    op[16 + i] = o1[i] * inv;
    op[32 + i] = o2[i] * inv;
    op[48 + i] = o3[i] * inv;
  }
}

// ---------------------------------------------------------------- launch
extern "C" void kernel_launch(void* const* d_in, const int* in_sizes, int n_in,
                              void* d_out, int out_size, void* d_ws, size_t ws_size,
                              hipStream_t stream) {
  const float* x    = (const float*)d_in[0];   // [8,1024,768]
  const float* W    = (const float*)d_in[1];   // [768,2304]
  const float* bias = (const float*)d_in[2];   // [2304]

  const size_t XN = (size_t)8 * 1024 * 768;     // 6291456
  const size_t WN = (size_t)768 * 2304;         // 1769472
  const size_t HN = (size_t)8 * 12 * 1024 * 64; // 6291456

  unsigned short* xb  = (unsigned short*)d_ws;
  unsigned short* wbT = xb + XN;
  unsigned short* Qb  = wbT + WN;
  unsigned short* Kb  = Qb + HN;
  unsigned short* Vt  = Kb + HN;                // ~51.4 MB total

  cvt_bf16<<<(int)((XN + 255) / 256), 256, 0, stream>>>(x, xb, (int)XN);
  cvt_w_transpose<<<dim3(24, 72), 256, 0, stream>>>(W, wbT);
  qkv_gemm<<<dim3(128, 9), 256, 0, stream>>>(xb, wbT, bias, Qb, Kb, Vt);
  mha_attn<<<dim3(16, 96), 128, 0, stream>>>(Qb, Kb, Vt, (float*)d_out);
}
